// InteractionBlock_69956427317729
// MI455X (gfx1250) — matrix operand order repaired and checked by reading, not compiled, not run on hardware
//
#include <hip/hip_runtime.h>
#include <cstdint>
#include <cstddef>

// ---------------------------------------------------------------------------
// CDNA5 (gfx1250) fused InteractionBlock.
//   B=2, N=512, NG=32, NF=NAB=128
// Core trick:  y = T @ U  with K = N*G = 16384,
//   T[i,(j,g)] = A[i,j]*exp(coeff[g]*(e[i,j]-off[g])^2)   (built in registers)
//   U[(j,g),f] = rf[j,f]*W_df2[g,f]                       (precomputed, bf16, L2-resident)
// All GEMMs use v_wmma_f32_16x16x32_bf16 (wave32, 16x16x32).
// ---------------------------------------------------------------------------

typedef __attribute__((ext_vector_type(16))) __bf16 v16bf;
typedef __attribute__((ext_vector_type(8)))  float  v8f;

#define KB   2
#define KN   512
#define KG   32
#define KF   128

union BFFrag { uint4 u[2]; v16bf v; };

__device__ __forceinline__ v16bf load_frag_g(const __bf16* p) {
  // 16 contiguous bf16 (32B) per lane, 16B aligned by construction.
  BFFrag f;
  f.u[0] = *reinterpret_cast<const uint4*>(p);
  f.u[1] = *reinterpret_cast<const uint4*>(p + 8);
  return f.v;
}

__device__ __forceinline__ v8f wmma_bf16(v16bf a, v16bf b, v8f c) {
  return __builtin_amdgcn_wmma_f32_16x16x32_bf16(false, a, false, b, (short)0, c,
                                                 false, false);
}

__device__ __forceinline__ float ssp_f(float x) {
  // ln(1+e^x) - ln2, overflow-safe
  return fmaxf(x, 0.0f) + __logf(1.0f + __expf(-fabsf(x))) - 0.69314718056f;
}

__device__ __forceinline__ uint32_t pack2(float lo, float hi) {
  __bf16 l = (__bf16)lo, h = (__bf16)hi;
  uint16_t lu = __builtin_bit_cast(uint16_t, l);
  uint16_t hu = __builtin_bit_cast(uint16_t, h);
  return (uint32_t)lu | ((uint32_t)hu << 16);
}

// ---------------------------------------------------------------------------
// Kernel 0: transpose+convert a 128x128 f32 weight to bf16 [n][k] layout.
// ---------------------------------------------------------------------------
__global__ void k_transpose(const float* __restrict__ in, __bf16* __restrict__ out) {
  int idx = blockIdx.x * 256 + threadIdx.x;      // 16384 total
  int k = idx >> 7, n = idx & 127;
  out[n * 128 + k] = (__bf16)in[idx];            // out[n][k] = in[k][n]
}

// ---------------------------------------------------------------------------
// Kernel 1: rf = r @ W_af (WMMA); emit rfT bf16 [b][f][j] and
//           U bf16 [b][j][f][g] = rf[b,j,f]*W_df2[g,f].
// Grid: 64 blocks (B*N/16), 256 threads (8 waves, wave w -> f-tile w).
// ---------------------------------------------------------------------------
__global__ __launch_bounds__(256) void k_rf_u(
    const float* __restrict__ r, const float* __restrict__ W_df2,
    const __bf16* __restrict__ WafT,
    __bf16* __restrict__ rfT, __bf16* __restrict__ U) {
  __shared__ float wL[KF * KG];                  // W_df2 transposed: [f][g], 16KB
  int tid = threadIdx.x;
  for (int idx = tid; idx < KF * KG; idx += 256) {
    int g = idx >> 7, f = idx & 127;
    wL[f * KG + g] = W_df2[idx];
  }
  __syncthreads();

  int tile = blockIdx.x;
  int b  = tile >> 5;
  int j0 = (tile & 31) * 16;
  int w = tid >> 5, lane = tid & 31, half = lane >> 4, il = lane & 15;
  int kb = half * 16;
  int f  = w * 16 + il;

  const float* rrow = r + ((size_t)(b * KN + j0 + il)) * KF;
  v8f acc = {0.f, 0.f, 0.f, 0.f, 0.f, 0.f, 0.f, 0.f};
#pragma unroll
  for (int kc = 0; kc < 4; ++kc) {
    v16bf a;
#pragma unroll
    for (int idx = 0; idx < 16; ++idx) {
      int k = kc * 32 + kb + idx;
      a[idx] = (__bf16)rrow[k];
    }
    v16bf bf = load_frag_g(WafT + f * 128 + kc * 32 + half * 16);
    acc = wmma_bf16(a, bf, acc);
  }

#pragma unroll
  for (int v = 0; v < 8; ++v) {
    int m = v + half * 8;
    int j = j0 + m;
    float val = acc[v];
    rfT[((size_t)(b * KF + f)) * KN + j] = (__bf16)val;
    uint32_t packed[16];
#pragma unroll
    for (int p = 0; p < 16; ++p)
      packed[p] = pack2(val * wL[f * KG + 2 * p], val * wL[f * KG + 2 * p + 1]);
    uint4* dst = reinterpret_cast<uint4*>(U + (((size_t)(b * KN + j)) * KF + f) * KG);
    dst[0] = make_uint4(packed[0],  packed[1],  packed[2],  packed[3]);
    dst[1] = make_uint4(packed[4],  packed[5],  packed[6],  packed[7]);
    dst[2] = make_uint4(packed[8],  packed[9],  packed[10], packed[11]);
    dst[3] = make_uint4(packed[12], packed[13], packed[14], packed[15]);
  }
}

// ---------------------------------------------------------------------------
// Kernel 2: main fused cfconv GEMM.  Grid: 64 i-tiles x 8 j-splits = 512 blocks.
// Each wave: 8 j's; per j build T fragment (16 exps/lane) and do 8 WMMAs
// against U.  Partials reduced via ds_add_f32 then global_atomic_add_f32.
// ---------------------------------------------------------------------------
__global__ __launch_bounds__(256) void k_main(
    const float* __restrict__ e, const float* __restrict__ A,
    const float* __restrict__ offsets, const float* __restrict__ widths,
    const __bf16* __restrict__ U, float* __restrict__ y_pre) {
  __shared__ float eL[16 * 64];
  __shared__ float aL[16 * 64];
  __shared__ float yL[16 * KF];
  int tid = threadIdx.x;
  int blk = blockIdx.x;
  int tile = blk >> 3;
  int s    = blk & 7;
  int b  = tile >> 5;
  int i0 = (tile & 31) * 16;
  int jbase = s * 64;

  for (int idx = tid; idx < 1024; idx += 256) {
    int row = idx >> 6, col = idx & 63;
    size_t g = ((size_t)(b * KN + i0 + row)) * KN + jbase + col;
    eL[idx] = e[g];
    aL[idx] = A[g];
  }
  for (int idx = tid; idx < 16 * KF; idx += 256) yL[idx] = 0.0f;
  __syncthreads();

  int w = tid >> 5, lane = tid & 31, half = lane >> 4, il = lane & 15;
  int kb = half * 16;

  float cg[16], og[16];
#pragma unroll
  for (int idx = 0; idx < 16; ++idx) {
    int g = kb + idx;
    float wd = widths[g];
    cg[idx] = -0.5f / (wd * wd);
    og[idx] = offsets[g];
  }

  v8f acc[8];
#pragma unroll
  for (int ft = 0; ft < 8; ++ft) acc[ft] = (v8f){0.f,0.f,0.f,0.f,0.f,0.f,0.f,0.f};

  for (int jj = 0; jj < 8; ++jj) {
    int jloc = w * 8 + jj;
    int j = jbase + jloc;
    float ev = eL[il * 64 + jloc];
    float av = aL[il * 64 + jloc];
    v16bf afrag;
#pragma unroll
    for (int idx = 0; idx < 16; ++idx) {
      float d = ev - og[idx];
      afrag[idx] = (__bf16)(av * __expf(cg[idx] * d * d));
    }
    const __bf16* ub = U + ((size_t)(b * KN + j)) * (KF * KG);
    __builtin_prefetch(ub + KF * KG, 0, 1);   // next j slice -> global_prefetch
#pragma unroll
    for (int ft = 0; ft < 8; ++ft) {
      v16bf bfrag = load_frag_g(ub + (ft * 16 + il) * KG + half * 16);
      acc[ft] = wmma_bf16(afrag, bfrag, acc[ft]);
    }
  }

  // cross-wave reduce in LDS (ds_add_f32)
#pragma unroll
  for (int ft = 0; ft < 8; ++ft)
#pragma unroll
    for (int v = 0; v < 8; ++v)
      atomicAdd(&yL[(v + half * 8) * KF + ft * 16 + il], acc[ft][v]);
  __syncthreads();

  for (int idx = tid; idx < 16 * KF; idx += 256)
    unsafeAtomicAdd(&y_pre[((size_t)(b * KN + i0 + (idx >> 7))) * KF + (idx & 127)],
                    yL[idx]);
}

// ---------------------------------------------------------------------------
// Kernel 3: bias GEMM (b_df2 * (A @ rfT)) + dense1 (+ssp) + dense2 -> out.
// Grid: 64 blocks (one per i-tile), 8 waves; wave w owns output tile w.
// ---------------------------------------------------------------------------
__global__ __launch_bounds__(256) void k_finish(
    const float* __restrict__ A, const float* __restrict__ y_pre,
    const __bf16* __restrict__ rfT, const float* __restrict__ b_df2,
    const __bf16* __restrict__ Wd1T, const float* __restrict__ b_d1,
    const __bf16* __restrict__ Wd2T, const float* __restrict__ b_d2,
    float* __restrict__ out) {
  __shared__ float  aL[16 * KN];    // 32KB
  __shared__ float  yL[16 * KF];    // 8KB
  __shared__ __bf16 y1L[16 * KF];   // 4KB
  int tid = threadIdx.x;
  int tile = blockIdx.x;
  int b  = tile >> 5;
  int i0 = (tile & 31) * 16;

  for (int idx = tid; idx < 16 * KN; idx += 256)
    aL[idx] = A[((size_t)(b * KN + i0 + (idx >> 9))) * KN + (idx & 511)];
  for (int idx = tid; idx < 16 * KF; idx += 256)
    yL[idx] = y_pre[((size_t)(b * KN + i0 + (idx >> 7))) * KF + (idx & 127)];
  __syncthreads();

  int w = tid >> 5, lane = tid & 31, half = lane >> 4, il = lane & 15;
  int kb = half * 16;
  int n  = w * 16 + il;

  // ---- bias term: y += b_df2[f] * (A(16x512) @ rfT) ----
  {
    v8f acc = {0.f,0.f,0.f,0.f,0.f,0.f,0.f,0.f};
    for (int kc = 0; kc < 16; ++kc) {
      v16bf a;
#pragma unroll
      for (int idx = 0; idx < 16; ++idx) {
        int k = kc * 32 + kb + idx;
        a[idx] = (__bf16)aL[il * KN + k];
      }
      v16bf bf = load_frag_g(rfT + ((size_t)(b * KF + n)) * KN + kc * 32 + half * 16);
      acc = wmma_bf16(a, bf, acc);
    }
    float bscale = b_df2[n];
#pragma unroll
    for (int v = 0; v < 8; ++v)
      yL[(v + half * 8) * KF + n] += bscale * acc[v];   // wave owns column n
  }
  __syncthreads();

  // ---- dense1: y1 = ssp(y @ W_d1 + b_d1) ----
  {
    v8f acc = {0.f,0.f,0.f,0.f,0.f,0.f,0.f,0.f};
#pragma unroll
    for (int kc = 0; kc < 4; ++kc) {
      v16bf a;
#pragma unroll
      for (int idx = 0; idx < 16; ++idx) {
        int k = kc * 32 + kb + idx;
        a[idx] = (__bf16)yL[il * KF + k];
      }
      v16bf bf = load_frag_g(Wd1T + n * 128 + kc * 32 + half * 16);
      acc = wmma_bf16(a, bf, acc);
    }
    float bias = b_d1[n];
#pragma unroll
    for (int v = 0; v < 8; ++v)
      y1L[(v + half * 8) * KF + n] = (__bf16)ssp_f(acc[v] + bias);
  }
  __syncthreads();

  // ---- dense2: out = y1 @ W_d2 + b_d2 ----
  {
    v8f acc = {0.f,0.f,0.f,0.f,0.f,0.f,0.f,0.f};
#pragma unroll
    for (int kc = 0; kc < 4; ++kc) {
      v16bf a;
#pragma unroll
      for (int idx = 0; idx < 16; ++idx) {
        int k = kc * 32 + kb + idx;
        a[idx] = y1L[il * KF + k];
      }
      v16bf bf = load_frag_g(Wd2T + n * 128 + kc * 32 + half * 16);
      acc = wmma_bf16(a, bf, acc);
    }
    float bias = b_d2[n];
#pragma unroll
    for (int v = 0; v < 8; ++v)
      out[((size_t)(b * KN + i0 + v + half * 8)) * KF + n] = acc[v] + bias;
  }
}

// ---------------------------------------------------------------------------
// Host launcher
// ---------------------------------------------------------------------------
extern "C" void kernel_launch(void* const* d_in, const int* in_sizes, int n_in,
                              void* d_out, int out_size, void* d_ws, size_t ws_size,
                              hipStream_t stream) {
  const float* r       = (const float*)d_in[0];
  const float* e       = (const float*)d_in[1];
  const float* A       = (const float*)d_in[2];
  const float* offsets = (const float*)d_in[3];
  const float* widths  = (const float*)d_in[4];
  // d_in[5] = W_df1, d_in[6] = b_df1: output discarded in reference -> skipped
  const float* W_df2   = (const float*)d_in[7];
  const float* b_df2   = (const float*)d_in[8];
  const float* W_af    = (const float*)d_in[9];
  const float* W_d1    = (const float*)d_in[10];
  const float* b_d1    = (const float*)d_in[11];
  const float* W_d2    = (const float*)d_in[12];
  const float* b_d2    = (const float*)d_in[13];
  float* out = (float*)d_out;

  // workspace carve (all sizes multiples of 256B)
  char* ws = (char*)d_ws;
  __bf16* WafT = (__bf16*)ws;  ws += 128 * 128 * 2;                  // 32KB
  __bf16* Wd1T = (__bf16*)ws;  ws += 128 * 128 * 2;                  // 32KB
  __bf16* Wd2T = (__bf16*)ws;  ws += 128 * 128 * 2;                  // 32KB
  __bf16* rfT  = (__bf16*)ws;  ws += (size_t)KB * KF * KN * 2;       // 256KB
  __bf16* U    = (__bf16*)ws;  ws += (size_t)KB * KN * KF * KG * 2;  // 8MB
  float*  y_pre = (float*)ws;  ws += (size_t)KB * KN * KF * 4;       // 512KB

  // 0) weight transposes -> bf16
  k_transpose<<<64, 256, 0, stream>>>(W_af, WafT);
  k_transpose<<<64, 256, 0, stream>>>(W_d1, Wd1T);
  k_transpose<<<64, 256, 0, stream>>>(W_d2, Wd2T);

  // 1) rf + U precompute
  k_rf_u<<<KB * KN / 16, 256, 0, stream>>>(r, W_df2, WafT, rfT, U);

  // 2) zero partial accumulator
  hipMemsetAsync(y_pre, 0, (size_t)KB * KN * KF * 4, stream);

  // 3) main fused cfconv GEMM
  k_main<<<(KB * KN / 16) * 8, 256, 0, stream>>>(e, A, offsets, widths, U, y_pre);

  // 4) bias GEMM + dense layers -> out
  k_finish<<<KB * KN / 16, 256, 0, stream>>>(A, y_pre, rfT, b_df2,
                                             Wd1T, b_d1, Wd2T, b_d2, out);
  (void)in_sizes; (void)n_in; (void)out_size; (void)ws_size;
}
